// StructureBuilder_50603304681507
// MI455X (gfx1250) — compile-verified, matrix operand order checked by reading, not run on hardware
//
#include <hip/hip_runtime.h>

typedef __attribute__((ext_vector_type(16))) _Float16 v16h;
typedef __attribute__((ext_vector_type(8)))  _Float16 v8h;
typedef __attribute__((ext_vector_type(8)))  float    v8f;

#define LSZ   4096   // H*W
#define CSZ   64
#define ASZ   128
#define NHD   8
#define FPAD  160    // hc*KK = 144 padded to multiple of 32
#define FREAL 144
#define TW    64     // k-tile width (columns per streamed tile)

union AF { v16h v; v8h run[2]; _Float16 e[16]; };
union CF { v8f  v; float e[8]; };

// raw LDS byte offset of a __shared__ object (for async-to-LDS dst operand)
__device__ __forceinline__ uint32_t lds_addr(const void* p) {
    return (uint32_t)(uintptr_t)(__attribute__((address_space(3))) const void*)p;
}
// CDNA5 async global->LDS copy, 16B per lane, tracked by ASYNCcnt
__device__ __forceinline__ void async_b128(uint32_t lds_dst, const void* src) {
    asm volatile("global_load_async_to_lds_b128 %0, %1, off"
                 :: "v"(lds_dst), "v"(src) : "memory");
}

__device__ __forceinline__ v8f wmma_f16(v16h a, v16h b, v8f c) {
    return __builtin_amdgcn_wmma_f32_16x16x32_f16(false, a, false, b, (short)0, c, false, false);
}

// ---------------------------------------------------------------------------
// 1) q/k projections: qraw = m * (Wq f) + bq ; kraw = (1-m) * (Wk f) + bk
// ---------------------------------------------------------------------------
__global__ __launch_bounds__(256)
void project_qk(const float* __restrict__ f, const float* __restrict__ masks,
                const float* __restrict__ Wq, const float* __restrict__ bq,
                const float* __restrict__ Wk, const float* __restrict__ bk,
                float* __restrict__ qraw, float* __restrict__ kraw)
{
    int i = blockIdx.x * 256 + threadIdx.x;   // ASZ*LSZ threads
    int a = i >> 12, l = i & 4095;
    float aq = 0.f, ak = 0.f;
    #pragma unroll 8
    for (int c = 0; c < CSZ; ++c) {
        float fv = f[((size_t)c << 12) + l];
        aq += Wq[a * CSZ + c] * fv;
        ak += Wk[a * CSZ + c] * fv;
    }
    float m = masks[l] > 0.5f ? 1.f : 0.f;
    qraw[i] = m * aq + bq[a];
    kraw[i] = (1.f - m) * ak + bk[a];
}

// ---------------------------------------------------------------------------
// 2) instance norm + relu over L per channel (block = one channel)
// ---------------------------------------------------------------------------
__global__ __launch_bounds__(256)
void inorm_relu(const float* __restrict__ src, float* __restrict__ dst)
{
    __shared__ float ssum[256], ssq[256];
    int ch = blockIdx.x;
    const float* x = src + ((size_t)ch << 12);
    float s = 0.f, q = 0.f;
    for (int l = threadIdx.x; l < LSZ; l += 256) { float v = x[l]; s += v; q += v * v; }
    ssum[threadIdx.x] = s; ssq[threadIdx.x] = q;
    __syncthreads();
    for (int off = 128; off > 0; off >>= 1) {
        if (threadIdx.x < off) {
            ssum[threadIdx.x] += ssum[threadIdx.x + off];
            ssq[threadIdx.x]  += ssq[threadIdx.x + off];
        }
        __syncthreads();
    }
    float mu  = ssum[0] * (1.0f / LSZ);
    float var = ssq[0]  * (1.0f / LSZ) - mu * mu;
    float rs  = rsqrtf(var + 1e-5f);
    float* y = dst + ((size_t)ch << 12);
    for (int l = threadIdx.x; l < LSZ; l += 256) {
        float v = (x[l] - mu) * rs;
        y[l] = v > 0.f ? v : 0.f;
    }
}

// ---------------------------------------------------------------------------
// 3) dilated KxK patch banks, TRANSPOSED feature-innermost layout:
//    qphT[head][l][FPAD], f = (a%16)*9 + kk, a = head*16 + f/9, pad f>=144 = 0
// ---------------------------------------------------------------------------
__global__ __launch_bounds__(256)
void build_patches_T(const float* __restrict__ q, const float* __restrict__ k,
                     _Float16* __restrict__ qphT, _Float16* __restrict__ kphT)
{
    int j = blockIdx.x * 256 + threadIdx.x;   // NHD*LSZ*FPAD threads
    int f   = j % FPAD;
    int rem = j / FPAD;
    int l   = rem & 4095;
    int n   = rem >> 12;
    _Float16 qv = (_Float16)0.f, kv = (_Float16)0.f;
    if (f < FREAL) {
        int cl = f / 9, kk = f % 9;
        int a = n * 16 + cl;
        int y = l >> 6, x = l & 63;
        int yy = y + (kk / 3) * 2 - 2, xx = x + (kk % 3) * 2 - 2;
        if (yy >= 0 && yy < 64 && xx >= 0 && xx < 64) {
            int t = (yy << 6) | xx;
            qv = (_Float16)q[((size_t)a << 12) + t];
            kv = (_Float16)k[((size_t)a << 12) + t];
        }
    }
    qphT[j] = qv; kphT[j] = kv;
}

// ---------------------------------------------------------------------------
// 4) vfold[c,l] = (1/9) * sum_kk patches(f_unmasked)[c,kk,l]   (f16)
// ---------------------------------------------------------------------------
__global__ __launch_bounds__(256)
void build_vfold(const float* __restrict__ f, const float* __restrict__ masks,
                 _Float16* __restrict__ vfold)
{
    int i = blockIdx.x * 256 + threadIdx.x;   // CSZ*LSZ threads
    int c = i >> 12, l = i & 4095;
    int y = l >> 6, x = l & 63;
    float s = 0.f;
    #pragma unroll
    for (int kk = 0; kk < 9; ++kk) {
        int yy = y + (kk / 3) * 2 - 2, xx = x + (kk % 3) * 2 - 2;
        if (yy >= 0 && yy < 64 && xx >= 0 && xx < 64) {
            int t = (yy << 6) | xx;
            float m = masks[t] > 0.5f ? 1.f : 0.f;
            s += f[((size_t)c << 12) + t] * (1.f - m);
        }
    }
    vfold[i] = (_Float16)(s * (1.0f / 9.0f));
}

__global__ __launch_bounds__(256)
void zero_f32(float* __restrict__ p, int n)
{
    int i = blockIdx.x * 256 + threadIdx.x;
    if (i < n) p[i] = 0.f;
}

// ---------------------------------------------------------------------------
// 5) flash attention: grid = 8 heads x 64 q-tiles, block = 128 (4 wave32)
//    each wave: 16 q-rows, streams 4096 k-cols in double-buffered 64-col tiles
//    per tile: 20x v_wmma (scores) + 8x v_wmma (value)
//    staging: global_load_async_to_lds_b128 pipelined one tile ahead
// ---------------------------------------------------------------------------
__global__ __launch_bounds__(128)
void attn_kernel(const _Float16* __restrict__ qphT,
                 const _Float16* __restrict__ kphT,
                 const _Float16* __restrict__ vfold,
                 float* __restrict__ attended)
{
    __shared__ __align__(16) _Float16 kp_lds[2][TW * FPAD];   // 2 x 20 KB, [j][f]
    __shared__ __align__(16) _Float16 v_lds[2][CSZ * TW];     // 2 x  8 KB, [c][j]
    __shared__ __align__(16) _Float16 p_lds[4][16 * TW];      //     8 KB, per-wave [m][j]

    const int tid   = threadIdx.x;
    const int lane  = tid & 31;
    const int wave  = tid >> 5;
    const int head  = blockIdx.x >> 6;
    const int qtile = blockIdx.x & 63;
    const int qbase = qtile * 64 + wave * 16;
    const int h     = lane >> 4;               // half-group
    const int n16   = lane & 15;

    const uint32_t kp_off[2] = { lds_addr(kp_lds[0]), lds_addr(kp_lds[1]) };
    const uint32_t v_off[2]  = { lds_addr(v_lds[0]),  lds_addr(v_lds[1])  };

    // issue one tile's async loads: kp tile = contiguous 20 KB, v tile = 64x128B
    auto stage = [&](int k0, int buf) {
        const char* kpg = (const char*)(kphT + (size_t)(head * LSZ + k0) * FPAD);
        #pragma unroll
        for (int i = 0; i < 10; ++i) {         // 1280 x 16B chunks
            int c16 = tid + i * 128;
            async_b128(kp_off[buf] + c16 * 16, kpg + c16 * 16);
        }
        #pragma unroll
        for (int i = 0; i < 4; ++i) {          // 512 x 16B chunks
            int c16  = tid + i * 128;
            int c    = c16 >> 3;
            int part = c16 & 7;
            const char* vg = (const char*)vfold + ((size_t)(c << 12) + k0) * 2 + part * 16;
            async_b128(v_off[buf] + c16 * 16, vg);
        }
    };

    // Q fragments: 5 chunks of K=32, resident for the whole k-loop.
    // A-frag runs: e0..7 -> k = h*8+e ; e8..15 -> k = 16+h*8+(e-8)
    const _Float16* qrow = qphT + ((size_t)(head * LSZ + qbase + n16) * FPAD);
    AF qf[5];
    #pragma unroll
    for (int fc = 0; fc < 5; ++fc) {
        qf[fc].run[0] = *(const v8h*)(qrow + fc * 32 + h * 8);
        qf[fc].run[1] = *(const v8h*)(qrow + fc * 32 + 16 + h * 8);
    }

    CF acc[4];
    #pragma unroll
    for (int cc = 0; cc < 4; ++cc)
        #pragma unroll
        for (int r = 0; r < 8; ++r) acc[cc].e[r] = 0.f;
    float rmax[8], rsum[8];
    #pragma unroll
    for (int r = 0; r < 8; ++r) { rmax[r] = -1e30f; rsum[r] = 0.f; }

    stage(0, 0);                                  // prologue: tile 0 in flight

    for (int k0 = 0; k0 < LSZ; k0 += TW) {
        const int buf = (k0 >> 6) & 1;
        if (k0 + TW < LSZ) {
            stage(k0 + TW, buf ^ 1);              // next tile overlaps compute
            asm volatile("s_wait_asynccnt 0xe" ::: "memory");   // drain current tile only
        } else {
            asm volatile("s_wait_asynccnt 0x0" ::: "memory");
        }
        __syncthreads();

        const _Float16* kb = kp_lds[buf];
        const _Float16* vb = v_lds[buf];

        // ---- scores: S[16x64] = Q(16x160) * Kp(160x64), 4 column groups ----
        CF sg[4];
        #pragma unroll
        for (int g = 0; g < 4; ++g) {
            const _Float16* kr = kb + (g * 16 + n16) * FPAD;
            #pragma unroll
            for (int r = 0; r < 8; ++r) sg[g].e[r] = 0.f;
            AF bcur;
            bcur.run[0] = *(const v8h*)(kr + h * 8);
            bcur.run[1] = *(const v8h*)(kr + 16 + h * 8);
            #pragma unroll
            for (int fc = 0; fc < 5; ++fc) {
                AF bnext;
                if (fc < 4) {                     // prefetch next frag before wmma
                    bnext.run[0] = *(const v8h*)(kr + (fc + 1) * 32 + h * 8);
                    bnext.run[1] = *(const v8h*)(kr + (fc + 1) * 32 + 16 + h * 8);
                }
                sg[g].v = wmma_f16(qf[fc].v, bcur.v, sg[g].v);
                if (fc < 4) bcur = bnext;
            }
        }

        // ---- streaming softmax update (rows r + 8h, cols across 16 lanes) ----
        float alpha[8];
        #pragma unroll
        for (int r = 0; r < 8; ++r) {
            float tm = fmaxf(fmaxf(sg[0].e[r], sg[1].e[r]), fmaxf(sg[2].e[r], sg[3].e[r]));
            #pragma unroll
            for (int m = 8; m >= 1; m >>= 1) tm = fmaxf(tm, __shfl_xor(tm, m, 32));
            float nm = fmaxf(rmax[r], tm);
            float a  = __expf(rmax[r] - nm);
            float ts = 0.f;
            #pragma unroll
            for (int g = 0; g < 4; ++g) {
                float p = __expf(sg[g].e[r] - nm);
                sg[g].e[r] = p;
                ts += p;
            }
            #pragma unroll
            for (int m = 8; m >= 1; m >>= 1) ts += __shfl_xor(ts, m, 32);
            rsum[r] = rsum[r] * a + ts;
            rmax[r] = nm;
            alpha[r] = a;
        }
        #pragma unroll
        for (int cc = 0; cc < 4; ++cc)
            #pragma unroll
            for (int r = 0; r < 8; ++r) acc[cc].e[r] *= alpha[r];

        // ---- transpose P (D-layout) -> A-layout via per-wave LDS ----
        _Float16* pw = p_lds[wave];
        #pragma unroll
        for (int g = 0; g < 4; ++g)
            #pragma unroll
            for (int r = 0; r < 8; ++r) {
                int m = r + (h << 3);
                pw[m * TW + g * 16 + n16] = (_Float16)sg[g].e[r];
            }
        asm volatile("s_wait_dscnt 0x0" ::: "memory");
        AF pA0, pA1;                              // P split into two K=32 A-frags
        pA0.run[0] = *(const v8h*)(pw + n16 * TW + h * 8);
        pA0.run[1] = *(const v8h*)(pw + n16 * TW + 16 + h * 8);
        pA1.run[0] = *(const v8h*)(pw + n16 * TW + 32 + h * 8);
        pA1.run[1] = *(const v8h*)(pw + n16 * TW + 32 + 16 + h * 8);

        // ---- value: out[16x64] += P(16x64) * V(64x64) ----
        #pragma unroll
        for (int cc = 0; cc < 4; ++cc) {
            const _Float16* vr = vb + (cc * 16 + n16) * TW;
            AF b0, b1;
            b0.run[0] = *(const v8h*)(vr + h * 8);
            b0.run[1] = *(const v8h*)(vr + 16 + h * 8);
            b1.run[0] = *(const v8h*)(vr + 32 + h * 8);
            b1.run[1] = *(const v8h*)(vr + 32 + 16 + h * 8);
            acc[cc].v = wmma_f16(pA0.v, b0.v, acc[cc].v);
            acc[cc].v = wmma_f16(pA1.v, b1.v, acc[cc].v);
        }
        __syncthreads();                          // tile fully consumed
    }

    // finalize: divide by softmax denom, fold 1/NH head mean, accumulate heads
    #pragma unroll
    for (int cc = 0; cc < 4; ++cc)
        #pragma unroll
        for (int r = 0; r < 8; ++r) {
            int row = qbase + r + (h << 3);
            int c   = cc * 16 + n16;
            atomicAdd(&attended[((size_t)c << 12) + row],
                      acc[cc].e[r] / rsum[r] * (1.0f / NHD));
        }
}

// ---------------------------------------------------------------------------
// 6) output projection: out1 = Wo * attended + bo
// ---------------------------------------------------------------------------
__global__ __launch_bounds__(256)
void project_out(const float* __restrict__ attended, const float* __restrict__ Wo,
                 const float* __restrict__ bo, float* __restrict__ out1)
{
    int i = blockIdx.x * 256 + threadIdx.x;   // CSZ*LSZ threads
    int o = i >> 12, l = i & 4095;
    float acc = bo[o];
    #pragma unroll 8
    for (int c = 0; c < CSZ; ++c)
        acc += Wo[o * CSZ + c] * attended[((size_t)c << 12) + l];
    out1[i] = acc;
}

// ---------------------------------------------------------------------------
extern "C" void kernel_launch(void* const* d_in, const int* in_sizes, int n_in,
                              void* d_out, int out_size, void* d_ws, size_t ws_size,
                              hipStream_t stream)
{
    (void)in_sizes; (void)n_in; (void)out_size; (void)ws_size;
    const float* features = (const float*)d_in[0];
    const float* masks    = (const float*)d_in[1];
    const float* Wq       = (const float*)d_in[2];
    const float* bq       = (const float*)d_in[3];
    const float* Wk       = (const float*)d_in[4];
    const float* bk       = (const float*)d_in[5];
    const float* Wo       = (const float*)d_in[6];
    const float* bo       = (const float*)d_in[7];
    float* out = (float*)d_out;

    char* ws = (char*)d_ws;
    size_t off = 0;
    auto alloc = [&](size_t bytes) -> void* {
        void* p = ws + off;
        off += (bytes + 255) & ~(size_t)255;
        return p;
    };
    float*     qkraw = (float*)alloc((size_t)2 * ASZ * LSZ * sizeof(float)); // q||k
    float*     qraw  = qkraw;
    float*     kraw  = qkraw + (size_t)ASZ * LSZ;
    _Float16*  qphT  = (_Float16*)alloc((size_t)NHD * LSZ * FPAD * 2);
    _Float16*  kphT  = (_Float16*)alloc((size_t)NHD * LSZ * FPAD * 2);
    _Float16*  vfold = (_Float16*)alloc((size_t)CSZ * LSZ * 2);
    float*     attnd = (float*)alloc((size_t)CSZ * LSZ * sizeof(float));
    float*     out1  = (float*)alloc((size_t)CSZ * LSZ * sizeof(float));

    project_qk<<<(ASZ * LSZ) / 256, 256, 0, stream>>>(features, masks, Wq, bq, Wk, bk, qraw, kraw);
    inorm_relu<<<2 * ASZ, 256, 0, stream>>>(qkraw, qkraw);                  // 256 channels in-place
    build_patches_T<<<(NHD * LSZ * FPAD) / 256, 256, 0, stream>>>(qraw, kraw, qphT, kphT);
    build_vfold<<<(CSZ * LSZ) / 256, 256, 0, stream>>>(features, masks, vfold);
    zero_f32<<<(CSZ * LSZ) / 256, 256, 0, stream>>>(attnd, CSZ * LSZ);
    attn_kernel<<<NHD * (LSZ / 64), 128, 0, stream>>>(qphT, kphT, vfold, attnd);
    project_out<<<(CSZ * LSZ) / 256, 256, 0, stream>>>(attnd, Wo, bo, out1);
    inorm_relu<<<CSZ, 256, 0, stream>>>(out1, out);
}
